// BSplineKANLayer_78176994722390
// MI455X (gfx1250) — matrix operand order repaired and checked by reading, not compiled, not run on hardware
//
#include <hip/hip_runtime.h>
#include <hip/hip_bf16.h>
#include <math.h>
#include <stdint.h>

// ---- problem constants (match reference) ----
#define BATCH     16384
#define IN_DIM    512
#define OUT_DIM   512
#define NC        11            // spline coefficients per (o,i)
#define NV        12            // 11 basis + 1 silu slot
#define NK        15            // knots per input dim
#define K_AUG     (IN_DIM*NV)   // 6144 augmented K
#define GROW      64            // grid_ws row stride (floats)

// ---- tiling ----
#define BM        64            // batch rows per block
#define BN        256           // out cols per block (recompute factor OUT/BN = 2)
#define KI        8             // input dims per K-chunk
#define KC        (KI*NV)       // 96 aug-K per chunk (3 WMMA K-steps)
#define KCP_A     96            // A LDS stride (192B; 4-way conflicts on only 12 loads/chunk)
#define KCP_B     102           // B LDS stride: 96 + TDM pad (1 DWORD per 64B) = 204B = 51 banks
                                //   -> odd bank stride, conflict-free fragment reads
#define NCHUNK    (IN_DIM/KI)   // 64

typedef _Float16 f16;
typedef __attribute__((ext_vector_type(16))) _Float16 v16h;
typedef __attribute__((ext_vector_type(8)))  _Float16 v8h;
typedef __attribute__((ext_vector_type(4)))  _Float16 h4;
typedef __attribute__((ext_vector_type(8)))  float    v8f;
typedef __attribute__((ext_vector_type(4)))  unsigned int u32x4;
typedef __attribute__((ext_vector_type(8)))  int          i32x8;
typedef __attribute__((ext_vector_type(4)))  int          i32x4;

// ---------------------------------------------------------------------------
// Kernel 1: knots + reciprocal Cox-de Boor denominators per input dim.
// grid_ws row (stride 64 floats): [0..14]=t, [16..29]=1/(t[j+1]-t[j]),
// [32..44]=1/(t[j+2]-t[j]), [48..59]=1/(t[j+3]-t[j])   (den==0 -> 1.0)
// ---------------------------------------------------------------------------
__global__ void kan_grid_kernel(const float* __restrict__ steps_log,
                                const float* __restrict__ grid_start,
                                float* __restrict__ grid_ws) {
    int i = blockIdx.x * blockDim.x + threadIdx.x;
    if (i >= IN_DIM) return;
    float t[NK];
    float acc = grid_start[i];
    t[0] = acc;
    #pragma unroll
    for (int j = 0; j < NK - 1; ++j) {
        float s  = steps_log[i * (NK - 1) + j];
        float sp = (s > 20.0f) ? s : log1pf(expf(s));   // softplus
        acc += sp;
        t[j + 1] = acc;
    }
    float* row = grid_ws + (size_t)i * GROW;
    #pragma unroll
    for (int j = 0; j < NK; ++j) row[j] = t[j];
    #pragma unroll
    for (int j = 0; j < NK - 1; ++j) {              // k=1: 14 values
        float den = t[j + 1] - t[j];
        row[16 + j] = (den == 0.f) ? 1.f : 1.f / den;
    }
    #pragma unroll
    for (int j = 0; j < NK - 2; ++j) {              // k=2: 13 values
        float den = t[j + 2] - t[j];
        row[32 + j] = (den == 0.f) ? 1.f : 1.f / den;
    }
    #pragma unroll
    for (int j = 0; j < NK - 3; ++j) {              // k=3: 12 values
        float den = t[j + 3] - t[j];
        row[48 + j] = (den == 0.f) ? 1.f : 1.f / den;
    }
}

// ---------------------------------------------------------------------------
// Kernel 2: pack augmented weights to f16: waug[o][i*12+c] = coeffs[o,i,c]
// (c<11) / base_weight[o,i] (c==11). One-time 11.5MB -> 6MB conversion.
// ---------------------------------------------------------------------------
__global__ void kan_pack_kernel(const float* __restrict__ coeffs,
                                const float* __restrict__ base_weight,
                                _Float16* __restrict__ waug) {
    int idx = blockIdx.x * blockDim.x + threadIdx.x;   // = o*512 + i
    if (idx >= OUT_DIM * IN_DIM) return;
    const float* cp = coeffs + (size_t)idx * NC;
    const float  bw = base_weight[idx];
    _Float16* dst = waug + (size_t)idx * NV;           // o*K_AUG + i*NV
    h4 v0 = { (f16)cp[0], (f16)cp[1], (f16)cp[2],  (f16)cp[3] };
    h4 v1 = { (f16)cp[4], (f16)cp[5], (f16)cp[6],  (f16)cp[7] };
    h4 v2 = { (f16)cp[8], (f16)cp[9], (f16)cp[10], (f16)bw    };
    *(h4*)(dst + 0) = v0;  *(h4*)(dst + 4) = v1;  *(h4*)(dst + 8) = v2;
}

// ---------------------------------------------------------------------------
// TDM: async-load one B tile (96 x 256 f16, k-contiguous per out-col) into LDS.
// pad_enable: +1 DWORD after every 64B stored -> LDS row stride 204B (51 banks)
// D# per CDNA5 ISA 8.3-8.5: 2D, data_size=2B, type=2 ("image").
// ---------------------------------------------------------------------------
__device__ __forceinline__ void tdm_load_B(unsigned int lds_off, uint64_t gaddr) {
    u32x4 g0;
    g0[0] = 1u;                                            // count=1, user mode
    g0[1] = lds_off;                                       // lds_addr (bytes)
    g0[2] = (unsigned int)(gaddr & 0xFFFFFFFFull);         // global_addr[31:0]
    g0[3] = (unsigned int)((gaddr >> 32) & 0x01FFFFFFull)  // global_addr[56:32]
          | (2u << 30);                                    // type=2
    i32x8 g1;
    g1[0] = (1 << 16)                                      // data_size=2B
          | (1 << 20)                                      // pad_enable
          | (3 << 22)                                      // pad_interval: 16 DWORDs (64B)
          | (0 << 25);                                     // pad_amount: 1 DWORD
    g1[1] = (int)((K_AUG & 0xFFFF) << 16);                 // tensor_dim0 lo16
    g1[2] = (int)((K_AUG >> 16) | (IN_DIM << 16));         // dim0 hi16 | tensor_dim1 lo16
    g1[3] = (int)((IN_DIM >> 16) | (KC << 16));            // dim1 hi16 | tile_dim0=96
    g1[4] = BN;                                            // tile_dim1=256, tile_dim2=0
    g1[5] = K_AUG;                                         // tensor_dim0_stride lo32
    g1[6] = 0;                                             // stride hi16 | dim1_stride lo16
    g1[7] = 0;
    i32x4 gz = { 0, 0, 0, 0 };                             // groups 2/3 unused (2D)
#if defined(__clang_major__) && (__clang_major__ >= 23)
    i32x8 gz8 = { 0, 0, 0, 0, 0, 0, 0, 0 };
    __builtin_amdgcn_tensor_load_to_lds(g0, g1, gz, gz, gz8, 0);
#else
    __builtin_amdgcn_tensor_load_to_lds(g0, g1, gz, gz, 0);
#endif
}

// ---------------------------------------------------------------------------
// Kernel 3: fused KAN layer: y = tanh( Aaug @ Waug + res_scale * x )
// ---------------------------------------------------------------------------
__global__ __launch_bounds__(256)
void BSplineKANLayer_wmma_kernel(const float* __restrict__ x,
                                 const _Float16* __restrict__ waug,
                                 const float* __restrict__ grid_ws,
                                 const float* __restrict__ res_scale,
                                 float* __restrict__ out) {
    __shared__ _Float16 ldsA[BM * KCP_A];   // 12.0 KB
    __shared__ _Float16 ldsB[BN * KCP_B];   // 51.0 KB (TDM destination, padded rows)

    const int tid    = threadIdx.x;
    const int lane   = tid & 31;
    const int wave   = tid >> 5;          // 0..7
    const int wm     = wave & 1;          // rows wm*32..+31
    const int wn     = wave >> 1;         // cols wn*64..+63

    const int bM0 = blockIdx.x * BM;
    const int oN0 = blockIdx.y * BN;

    const int i_loc  = tid & 7;           // fixed input-dim slot per thread
    const int rquart = tid >> 3;          // 0..31

    v8f acc[2][4];
    #pragma unroll
    for (int tm = 0; tm < 2; ++tm)
        #pragma unroll
        for (int tn = 0; tn < 4; ++tn)
            acc[tm][tn] = (v8f){0.f,0.f,0.f,0.f,0.f,0.f,0.f,0.f};

    const int arow0 = wm * 32 + (lane & 15);
    const int aksel = (lane >> 4) * 16;
    const int bcol0 = wn * 64 + (lane & 15);
    const int bksel = (lane >> 4) * 16;

    const unsigned int ldsB_off = (unsigned int)(size_t)(&ldsB[0]);
    const uint64_t waug_base = (uint64_t)(size_t)waug + (uint64_t)oN0 * K_AUG * 2;

    for (int ch = 0; ch < NCHUNK; ++ch) {
        const int ig = ch * KI + i_loc;
        const float* grow = grid_ws + (size_t)ig * GROW;

        __syncthreads();   // previous chunk's fragment reads complete

        // ---- kick off async B-tile DMA (wave 0 only; EXEC-independent op) ----
        if (tid < 32) {
            tdm_load_B(ldsB_off, waug_base + (uint64_t)(ch * KC) * 2);
        }

        // speculative prefetch of next chunk's x column (global_prefetch_b8)
        __builtin_prefetch(&x[(size_t)(bM0 + rquart) * IN_DIM + ig + KI], 0, 1);

        // ---- build A tile (overlaps the TDM transfer) ----
        float t[NK], i1[14], i2[13], i3[12];
        #pragma unroll
        for (int j = 0; j < NK; ++j) t[j]  = grow[j];
        #pragma unroll
        for (int j = 0; j < 14; ++j) i1[j] = grow[16 + j];
        #pragma unroll
        for (int j = 0; j < 13; ++j) i2[j] = grow[32 + j];
        #pragma unroll
        for (int j = 0; j < 12; ++j) i3[j] = grow[48 + j];

        #pragma unroll
        for (int p = 0; p < 2; ++p) {
            const int brow = rquart + 32 * p;                      // 0..63
            const float xv = x[(size_t)(bM0 + brow) * IN_DIM + ig];

            float d[NK];
            #pragma unroll
            for (int j = 0; j < NK; ++j) d[j] = xv - t[j];

            float bas[14];
            #pragma unroll
            for (int j = 0; j < 14; ++j)
                bas[j] = (d[j] >= 0.f && d[j + 1] < 0.f) ? 1.f : 0.f;
            #pragma unroll
            for (int j = 0; j < 13; ++j)   // k=1
                bas[j] = (d[j]*i1[j])*bas[j] - (d[j+2]*i1[j+1])*bas[j+1];
            #pragma unroll
            for (int j = 0; j < 12; ++j)   // k=2
                bas[j] = (d[j]*i2[j])*bas[j] - (d[j+3]*i2[j+1])*bas[j+1];
            #pragma unroll
            for (int j = 0; j < 11; ++j)   // k=3
                bas[j] = (d[j]*i3[j])*bas[j] - (d[j+4]*i3[j+1])*bas[j+1];

            const float sil = xv / (1.0f + expf(-xv));             // silu slot

            _Float16* dst = &ldsA[brow * KCP_A + i_loc * NV];
            h4 v0 = { (f16)bas[0], (f16)bas[1], (f16)bas[2],  (f16)bas[3] };
            h4 v1 = { (f16)bas[4], (f16)bas[5], (f16)bas[6],  (f16)bas[7] };
            h4 v2 = { (f16)bas[8], (f16)bas[9], (f16)bas[10], (f16)sil    };
            *(h4*)(dst + 0) = v0;  *(h4*)(dst + 4) = v1;  *(h4*)(dst + 8) = v2;
        }

        // ---- wait for DMA, then sync whole block ----
        if (tid < 32) __builtin_amdgcn_s_wait_tensorcnt(0);
        __syncthreads();

        // ---- 3 WMMA K-steps over the 96-wide chunk ----
        #pragma unroll
        for (int ks = 0; ks < 3; ++ks) {
            const int ka = ks * 32 + aksel;
            // B element k lives at padded f16 offset k + 2*(k>>5)
            const int kb  = ks * 32 + bksel;
            const int kbo = kb + 2 * ks;               // (kb>>5) == ks for bksel<32

            v16h afrag[2];
            #pragma unroll
            for (int tm = 0; tm < 2; ++tm) {
                const _Float16* ap = &ldsA[(arow0 + tm * 16) * KCP_A + ka];
                v8h lo = *(const v8h*)(ap);
                v8h hi = *(const v8h*)(ap + 8);
                afrag[tm] = __builtin_shufflevector(lo, hi,
                    0,1,2,3,4,5,6,7,8,9,10,11,12,13,14,15);
            }
            #pragma unroll
            for (int tn = 0; tn < 4; ++tn) {
                // conflict-free: 204B col stride = 51 banks (odd); 4B-aligned -> b32 loads
                const int* bp = (const int*)&ldsB[(bcol0 + tn * 16) * KCP_B + kbo];
                i32x8 bw_;
                #pragma unroll
                for (int q = 0; q < 8; ++q) bw_[q] = bp[q];
                v16h bfrag = __builtin_bit_cast(v16h, bw_);
                #pragma unroll
                for (int tm = 0; tm < 2; ++tm) {
                    acc[tm][tn] = __builtin_amdgcn_wmma_f32_16x16x32_f16(
                        false, afrag[tm], false, bfrag,
                        (short)0, acc[tm][tn], false, false);
                }
            }
        }
    }

    // ---- epilogue: y = tanh(acc + res_scale * x) ----
    const float rs = res_scale[0];
    #pragma unroll
    for (int tm = 0; tm < 2; ++tm) {
        #pragma unroll
        for (int tn = 0; tn < 4; ++tn) {
            const int col  = oN0 + wn * 64 + tn * 16 + (lane & 15);
            const int row0 = bM0 + wm * 32 + tm * 16 + ((lane >> 4) * 8);
            #pragma unroll
            for (int v = 0; v < 8; ++v) {
                const int row = row0 + v;
                const float xr = x[(size_t)row * IN_DIM + col];
                out[(size_t)row * OUT_DIM + col] = tanhf(acc[tm][tn][v] + rs * xr);
            }
        }
    }
}

// ---------------------------------------------------------------------------
extern "C" void kernel_launch(void* const* d_in, const int* in_sizes, int n_in,
                              void* d_out, int out_size, void* d_ws, size_t ws_size,
                              hipStream_t stream) {
    const float* x              = (const float*)d_in[0];
    const float* coeffs         = (const float*)d_in[1];
    const float* base_weight    = (const float*)d_in[2];
    const float* grid_steps_log = (const float*)d_in[3];
    const float* grid_start     = (const float*)d_in[4];
    const float* res_scale      = (const float*)d_in[5];
    float*       out            = (float*)d_out;

    float*    grid_ws = (float*)d_ws;                          // 512*64*4 = 128 KB
    _Float16* waug    = (_Float16*)((char*)d_ws + (size_t)IN_DIM * GROW * 4); // 6 MB

    kan_grid_kernel<<<(IN_DIM + 255) / 256, 256, 0, stream>>>(
        grid_steps_log, grid_start, grid_ws);
    kan_pack_kernel<<<(OUT_DIM * IN_DIM + 255) / 256, 256, 0, stream>>>(
        coeffs, base_weight, waug);

    dim3 grid(BATCH / BM, OUT_DIM / BN);   // 256 x 2 = 512 workgroups
    BSplineKANLayer_wmma_kernel<<<grid, 256, 0, stream>>>(
        x, waug, grid_ws, res_scale, out);
}
